// LlamaAttention_37933151158659
// MI455X (gfx1250) — compile-verified, matrix operand order checked by reading, not run on hardware
//
#include <hip/hip_runtime.h>
#include <hip/hip_bf16.h>

#define T_SEQ   2048
#define HIDDEN  4096
#define NHEADS  32
#define NKV     8
#define HDIM    128

typedef __attribute__((ext_vector_type(16))) __bf16 v16bf;
typedef __attribute__((ext_vector_type(8)))  float  v8f;
typedef int vi4 __attribute__((vector_size(16)));          // <4 x i32> payload type
typedef __attribute__((address_space(1))) vi4 gvi4;        // global 128-bit chunk
typedef __attribute__((address_space(3))) vi4 lvi4;        // LDS 128-bit chunk

#if defined(__has_builtin)
#if __has_builtin(__builtin_amdgcn_global_load_async_to_lds_b128)
#define HAVE_ASYNC_LDS 1
#endif
#endif

__device__ __forceinline__ unsigned short f2bf(float x) {
    unsigned u = __float_as_uint(x);
    unsigned r = (u + 0x7FFFu + ((u >> 16) & 1u)) >> 16;
    return (unsigned short)r;
}
__device__ __forceinline__ unsigned pack2(float a, float b) {
    return (unsigned)f2bf(a) | ((unsigned)f2bf(b) << 16);
}
__device__ __forceinline__ v8f zero8() {
    v8f z = {0.f,0.f,0.f,0.f,0.f,0.f,0.f,0.f};
    return z;
}
// One 16x32 bf16 operand fragment from a 32-element (64B) bf16 row:
// two b128 loads (LDS rows -> ds_load_b128, global rows -> global_load_b128).
__device__ __forceinline__ v16bf ld_frag(const unsigned short* rowbase, int half) {
    union { v16bf v; uint4 u[2]; } f;
    f.u[0] = *(const uint4*)(rowbase + half * 8);
    f.u[1] = *(const uint4*)(rowbase + 16 + half * 8);
    return f.v;
}
__device__ __forceinline__ v8f wmma_bf16(v16bf a, v16bf b, v8f c) {
    return __builtin_amdgcn_wmma_f32_16x16x32_bf16(false, a, false, b, (short)0, c, false, false);
}
// 16-byte global->LDS copy: async DMA (ASYNCcnt) on CDNA5, sync fallback else.
__device__ __forceinline__ void cp_g2l_b128(const unsigned short* g, unsigned short* l) {
#if HAVE_ASYNC_LDS
    __builtin_amdgcn_global_load_async_to_lds_b128((gvi4*)g, (lvi4*)l, 0, 0);
#else
    *(uint4*)l = *(const uint4*)g;
#endif
}
#if HAVE_ASYNC_LDS
#if __has_builtin(__builtin_amdgcn_s_wait_asynccnt)
#define WAIT_ASYNC(n) __builtin_amdgcn_s_wait_asynccnt(n)
#else
#define WAIT_ASYNC(n) __asm__ volatile("s_wait_asynccnt %0" :: "i"(n) : "memory")
#endif
#else
#define WAIT_ASYNC(n)
#endif

// -------------------------------------------------------------------------
// Elementwise f32 -> bf16 (vectorized).
// -------------------------------------------------------------------------
__global__ void __launch_bounds__(256)
cvt_bf16(const float* __restrict__ in, unsigned short* __restrict__ out) {
    long i = ((long)blockIdx.x * 256 + threadIdx.x) * 4;
    float4 x = *(const float4*)(in + i);
    uint2 pk;
    pk.x = pack2(x.x, x.y);
    pk.y = pack2(x.z, x.w);
    *(uint2*)(out + i) = pk;
}

// -------------------------------------------------------------------------
// Tiled transpose with conversion: in[K,N] f32 -> out[N,K] bf16 (k-contig).
// Block (32,8), tile 32x32 via LDS, coalesced both sides.
// -------------------------------------------------------------------------
__global__ void __launch_bounds__(256)
transpose_to_bf16(const float* __restrict__ in, unsigned short* __restrict__ out,
                  int K, int N) {
    __shared__ float t[32][33];
    const int n0 = blockIdx.x * 32, k0 = blockIdx.y * 32;
    const int tx = threadIdx.x, ty = threadIdx.y;
#pragma unroll
    for (int i = 0; i < 4; ++i)
        t[ty + i * 8][tx] = in[(long)(k0 + ty + i * 8) * N + n0 + tx];
    __syncthreads();
#pragma unroll
    for (int i = 0; i < 4; ++i)
        out[(long)(n0 + ty + i * 8) * K + k0 + tx] = f2bf(t[tx][ty + i * 8]);
}

// -------------------------------------------------------------------------
// GEMM: C[M,N] f32 = A[M,K] * BT[N,K]^T, A/BT bf16 k-contiguous.
// 128x128 block tile, K-step 32, 8 waves (wave tile 32x64, 8 WMMA/step).
// Double-buffered LDS staging via async global->LDS b128 copies.
// -------------------------------------------------------------------------
__global__ void __launch_bounds__(256)
gemm_bf16(const unsigned short* __restrict__ A, const unsigned short* __restrict__ BT,
          float* __restrict__ C, int M, int N, int K) {
    __shared__ __attribute__((aligned(16))) unsigned short As[2][128 * 32]; // [m][k]
    __shared__ __attribute__((aligned(16))) unsigned short Bs[2][128 * 32]; // [n][k]

    const int tid  = threadIdx.x;
    const int wid  = tid >> 5;
    const int lane = tid & 31;
    const int half = lane >> 4;
    const int ln   = lane & 15;
    const int wm   = wid >> 1;
    const int wn   = wid & 1;
    const int m0   = blockIdx.y * 128;
    const int n0   = blockIdx.x * 128;

    v8f acc[2][4];
#pragma unroll
    for (int i = 0; i < 2; ++i)
#pragma unroll
        for (int j = 0; j < 4; ++j) acc[i][j] = zero8();

    // Stage one 128x32 bf16 tile pair: 512 16B chunks each, 2 per thread,
    // 4 async copies per thread per stage.
    auto stage = [&](int buf, int k0) {
#pragma unroll
        for (int p = 0; p < 2; ++p) {
            int c   = tid + p * 256;
            int row = c >> 2;
            int seg = c & 3;
            cp_g2l_b128(A  + (long)(m0 + row) * K + k0 + seg * 8,
                        &As[buf][row * 32 + seg * 8]);
            cp_g2l_b128(BT + (long)(n0 + row) * K + k0 + seg * 8,
                        &Bs[buf][row * 32 + seg * 8]);
        }
    };

    const int NT = K >> 5;
    stage(0, 0);
    for (int kt = 0; kt < NT; ++kt) {
        const int cur = kt & 1;
        const bool pre = (kt + 1 < NT);
        if (pre) stage(cur ^ 1, (kt + 1) << 5);
        if (pre) { WAIT_ASYNC(4); } else { WAIT_ASYNC(0); }
        __syncthreads();

        v16bf afr[2];
#pragma unroll
        for (int i = 0; i < 2; ++i)
            afr[i] = ld_frag(&As[cur][(wm * 32 + i * 16 + ln) * 32], half);
#pragma unroll
        for (int j = 0; j < 4; ++j) {
            v16bf bfr = ld_frag(&Bs[cur][(wn * 64 + j * 16 + ln) * 32], half);
#pragma unroll
            for (int i = 0; i < 2; ++i)
                acc[i][j] = wmma_bf16(afr[i], bfr, acc[i][j]);
        }
        __syncthreads();  // protect buffer reuse before next prefetch lands
    }

#pragma unroll
    for (int i = 0; i < 2; ++i)
#pragma unroll
        for (int j = 0; j < 4; ++j) {
            float* cb = C + (long)(m0 + wm * 32 + i * 16 + half * 8) * N +
                        n0 + wn * 64 + j * 16 + ln;
#pragma unroll
            for (int r = 0; r < 8; ++r) cb[(long)r * N] = acc[i][j][r];
        }
}

// -------------------------------------------------------------------------
// NeoX RoPE + layout conversion. h<32: q -> pre-scaled bf16 q_bf.
// 32<=h<40: k -> bf16 k_bf + f32 kv_fused. 40<=h<48: v -> f32 kv_fused +
// transposed bf16 vT_bf[h][d][t].
// -------------------------------------------------------------------------
__global__ void __launch_bounds__(256)
rope_pack(const int* __restrict__ positions, const float* __restrict__ q,
          const float* __restrict__ k, const float* __restrict__ v,
          unsigned short* __restrict__ q_bf, unsigned short* __restrict__ k_bf,
          unsigned short* __restrict__ vT_bf, float* __restrict__ kv_out) {
    int idx = blockIdx.x * blockDim.x + threadIdx.x;
    int t   = idx / (48 * 64);
    int rem = idx - t * (48 * 64);
    int h   = rem >> 6;
    int j   = rem & 63;
    if (t >= T_SEQ) return;
    const float scale = 0.08838834764831845f;  // HDIM^-0.5 folded into q

    if (h < 40) {
        float pos = (float)positions[t];
        float fr  = pos * __expf(-(float)j * (2.0f / 128.0f) * 9.210340371976184f);
        float c, s;
        __sincosf(fr, &s, &c);
        if (h < 32) {
            const float* b = q + (long)t * (NHEADS * HDIM) + h * HDIM;
            float x1 = b[j], x2 = b[j + 64];
            unsigned short* o = q_bf + (long)t * (NHEADS * HDIM) + h * HDIM;
            o[j]      = f2bf((x1 * c - x2 * s) * scale);
            o[j + 64] = f2bf((x2 * c + x1 * s) * scale);
        } else {
            int kvh = h - 32;
            const float* b = k + (long)t * (NKV * HDIM) + kvh * HDIM;
            float x1 = b[j], x2 = b[j + 64];
            float o1 = x1 * c - x2 * s;
            float o2 = x2 * c + x1 * s;
            unsigned short* o = k_bf + (long)t * (NKV * HDIM) + kvh * HDIM;
            o[j]      = f2bf(o1);
            o[j + 64] = f2bf(o2);
            float* ko = kv_out + (long)t * (NKV * HDIM) + kvh * HDIM;
            ko[j]      = o1;
            ko[j + 64] = o2;
        }
    } else {
        int kvh = h - 40;
        const float* vb = v + (long)t * (NKV * HDIM) + kvh * HDIM;
        float x1 = vb[j], x2 = vb[j + 64];
        float* vo = kv_out + (long)(T_SEQ * NKV * HDIM) +
                    (long)t * (NKV * HDIM) + kvh * HDIM;
        vo[j]      = x1;
        vo[j + 64] = x2;
        unsigned short* vt = vT_bf + (long)kvh * (HDIM * T_SEQ);
        vt[(long)j * T_SEQ + t]        = f2bf(x1);
        vt[(long)(j + 64) * T_SEQ + t] = f2bf(x2);
    }
}

// -------------------------------------------------------------------------
// Flash-style causal GQA attention, all-bf16 operands. Grid (T/128, NHEADS),
// 8 waves x 16 query rows. K/V tiles async-copied to LDS (no conversion):
// K rows are k-contig in k_bf, V rows are key-contig in vT_bf. Per key
// block of 32: S = 8 WMMA, online softmax, P via per-wave LDS scratch,
// P*V = 8 WMMA. Output written as bf16 (feeds output-projection WMMA).
// -------------------------------------------------------------------------
__global__ void __launch_bounds__(256)
flash_attn(const unsigned short* __restrict__ q_bf,
           const unsigned short* __restrict__ k_bf,
           const unsigned short* __restrict__ vT_bf,
           unsigned short* __restrict__ attn) {
    __shared__ __attribute__((aligned(16))) unsigned short Ks[32 * 128];   // [key][d]
    __shared__ __attribute__((aligned(16))) unsigned short Vs[128 * 32];   // [d][key]
    __shared__ __attribute__((aligned(16))) unsigned short Ps[8][16 * 32]; // per-wave P

    const int tid  = threadIdx.x;
    const int wid  = tid >> 5;
    const int lane = tid & 31;
    const int half = lane >> 4;
    const int ln   = lane & 15;

    const int qt    = blockIdx.x;
    const int hq    = blockIdx.y;
    const int kvh   = hq >> 2;
    const int qrow0 = qt * 128 + wid * 16;

    // Q fragments straight from pre-scaled bf16 (global b128 pairs).
    v16bf qf[4];
#pragma unroll
    for (int c = 0; c < 4; ++c)
        qf[c] = ld_frag(q_bf + (long)(qrow0 + ln) * (NHEADS * HDIM) +
                        hq * HDIM + c * 32, half);

    v8f o[8];
#pragma unroll
    for (int dt = 0; dt < 8; ++dt) o[dt] = zero8();
    float mx[8], lsum[8];
#pragma unroll
    for (int r = 0; r < 8; ++r) { mx[r] = -3.0e38f; lsum[r] = 0.0f; }

    const int nkb = qt * 4 + 4;
    for (int kb = 0; kb < nkb; ++kb) {
        const int kbase = kb * 32;
        __syncthreads();  // previous iteration finished reading Ks/Vs
        // K tile: 32 keys x 256B -> 512 chunks; V tile: 128 d x 64B -> 512.
#pragma unroll
        for (int p = 0; p < 2; ++p) {
            int c = tid + p * 256;
            int key = c >> 4, segk = c & 15;
            cp_g2l_b128(k_bf + (long)(kbase + key) * (NKV * HDIM) + kvh * HDIM + segk * 8,
                        &Ks[key * 128 + segk * 8]);
            int d = c >> 2, segv = c & 3;
            cp_g2l_b128(vT_bf + (long)kvh * (HDIM * T_SEQ) + (long)d * T_SEQ +
                            kbase + segv * 8,
                        &Vs[d * 32 + segv * 8]);
        }
        WAIT_ASYNC(0);
        __syncthreads();

        // S = Q * K^T
        v8f s[2];
        s[0] = zero8(); s[1] = zero8();
#pragma unroll
        for (int j = 0; j < 2; ++j)
#pragma unroll
            for (int c = 0; c < 4; ++c) {
                v16bf bfr = ld_frag(&Ks[(j * 16 + ln) * 128 + c * 32], half);
                s[j] = wmma_bf16(qf[c], bfr, s[j]);
            }

        // Causal mask + online softmax; element (j,r) = (q row qrow0+r+8*half,
        // key kbase+j*16+ln); row reductions across the 16-lane half.
        float nmax[8];
#pragma unroll
        for (int r = 0; r < 8; ++r) {
            int qidx = qrow0 + r + 8 * half;
#pragma unroll
            for (int j = 0; j < 2; ++j)
                if (kbase + j * 16 + ln > qidx) s[j][r] = -3.0e38f;
            float mv = fmaxf(s[0][r], s[1][r]);
            mv = fmaxf(mv, __shfl_xor(mv, 1, 32));
            mv = fmaxf(mv, __shfl_xor(mv, 2, 32));
            mv = fmaxf(mv, __shfl_xor(mv, 4, 32));
            mv = fmaxf(mv, __shfl_xor(mv, 8, 32));
            nmax[r] = fmaxf(mx[r], mv);
        }
#pragma unroll
        for (int r = 0; r < 8; ++r) {
            float alpha = __expf(mx[r] - nmax[r]);
            mx[r] = nmax[r];
            float rs = 0.0f;
#pragma unroll
            for (int j = 0; j < 2; ++j) {
                float pv = __expf(s[j][r] - nmax[r]);
                s[j][r] = pv;
                rs += pv;
            }
            rs += __shfl_xor(rs, 1, 32);
            rs += __shfl_xor(rs, 2, 32);
            rs += __shfl_xor(rs, 4, 32);
            rs += __shfl_xor(rs, 8, 32);
            lsum[r] = lsum[r] * alpha + rs;
#pragma unroll
            for (int dt = 0; dt < 8; ++dt) o[dt][r] *= alpha;
        }

        // P: C-layout -> A-layout through wave-private LDS (in-order DS).
        unsigned short* ps = &Ps[wid][0];
#pragma unroll
        for (int j = 0; j < 2; ++j)
#pragma unroll
            for (int r = 0; r < 8; ++r)
                ps[(r + 8 * half) * 32 + j * 16 + ln] = f2bf(s[j][r]);
        __asm__ volatile("s_wait_dscnt 0" ::: "memory");
        v16bf pf = ld_frag(&ps[ln * 32], half);

        // O += P * V
#pragma unroll
        for (int dt = 0; dt < 8; ++dt) {
            v16bf vf = ld_frag(&Vs[(dt * 16 + ln) * 32], half);
            o[dt] = wmma_bf16(pf, vf, o[dt]);
        }
    }

    // Normalize, write attn as bf16 (multiplicand of output projection).
#pragma unroll
    for (int r = 0; r < 8; ++r) {
        float inv = 1.0f / lsum[r];
        unsigned short* ab = attn + (long)(qrow0 + r + 8 * half) * (NHEADS * HDIM) +
                             hq * HDIM + ln;
#pragma unroll
        for (int dt = 0; dt < 8; ++dt) ab[dt * 16] = f2bf(o[dt][r] * inv);
    }
}

// -------------------------------------------------------------------------
extern "C" void kernel_launch(void* const* d_in, const int* in_sizes, int n_in,
                              void* d_out, int out_size, void* d_ws, size_t ws_size,
                              hipStream_t stream) {
    const int*   positions = (const int*)  d_in[0];
    const float* hidden    = (const float*)d_in[1];
    const float* wq        = (const float*)d_in[2];
    const float* wk        = (const float*)d_in[3];
    const float* wv        = (const float*)d_in[4];
    const float* wo        = (const float*)d_in[5];

    // Workspace layout.
    float* qf32 = (float*)d_ws;                       // 2048*4096
    float* kf32 = qf32 + (long)T_SEQ * NHEADS * HDIM; // 2048*1024
    float* vf32 = kf32 + (long)T_SEQ * NKV * HDIM;    // 2048*1024
    unsigned short* hbf    = (unsigned short*)(vf32 + (long)T_SEQ * NKV * HDIM);
    unsigned short* wqT    = hbf + (long)T_SEQ * HIDDEN;
    unsigned short* wkT    = wqT + (long)HIDDEN * NHEADS * HDIM;
    unsigned short* wvT    = wkT + (long)HIDDEN * NKV * HDIM;
    unsigned short* woT    = wvT + (long)HIDDEN * NKV * HDIM;
    unsigned short* qbf    = woT + (long)NHEADS * HDIM * HIDDEN;
    unsigned short* kbf    = qbf + (long)T_SEQ * NHEADS * HDIM;
    unsigned short* vTbf   = kbf + (long)T_SEQ * NKV * HDIM;
    unsigned short* attnbf = vTbf + (long)T_SEQ * NKV * HDIM;

    float* out    = (float*)d_out;
    float* kv_out = out + (long)T_SEQ * NHEADS * HDIM;

    dim3 blk(256);
    // One-shot bf16 conversion / weight transposes (k-contiguous layouts).
    cvt_bf16<<<(T_SEQ * HIDDEN) / 1024, blk, 0, stream>>>(hidden, hbf);
    transpose_to_bf16<<<dim3(NHEADS * HDIM / 32, HIDDEN / 32), dim3(32, 8), 0, stream>>>(
        wq, wqT, HIDDEN, NHEADS * HDIM);
    transpose_to_bf16<<<dim3(NKV * HDIM / 32, HIDDEN / 32), dim3(32, 8), 0, stream>>>(
        wk, wkT, HIDDEN, NKV * HDIM);
    transpose_to_bf16<<<dim3(NKV * HDIM / 32, HIDDEN / 32), dim3(32, 8), 0, stream>>>(
        wv, wvT, HIDDEN, NKV * HDIM);
    transpose_to_bf16<<<dim3(HIDDEN / 32, NHEADS * HDIM / 32), dim3(32, 8), 0, stream>>>(
        wo, woT, NHEADS * HDIM, HIDDEN);
    // QKV projections (bf16 WMMA, async double-buffered staging).
    gemm_bf16<<<dim3(NHEADS * HDIM / 128, T_SEQ / 128), blk, 0, stream>>>(
        hbf, wqT, qf32, T_SEQ, NHEADS * HDIM, HIDDEN);
    gemm_bf16<<<dim3(NKV * HDIM / 128, T_SEQ / 128), blk, 0, stream>>>(
        hbf, wkT, kf32, T_SEQ, NKV * HDIM, HIDDEN);
    gemm_bf16<<<dim3(NKV * HDIM / 128, T_SEQ / 128), blk, 0, stream>>>(
        hbf, wvT, vf32, T_SEQ, NKV * HDIM, HIDDEN);
    // RoPE + bf16/transposed packing + kv_fused output.
    rope_pack<<<(T_SEQ * 48 * 64) / 256, blk, 0, stream>>>(
        positions, qf32, kf32, vf32, qbf, kbf, vTbf, kv_out);
    // Attention.
    flash_attn<<<dim3(T_SEQ / 128, NHEADS), blk, 0, stream>>>(qbf, kbf, vTbf, attnbf);
    // Output projection.
    gemm_bf16<<<dim3(HIDDEN / 128, T_SEQ / 128), blk, 0, stream>>>(
        attnbf, woT, out, T_SEQ, HIDDEN, HIDDEN);
}
